// TuckER_46600395162308
// MI455X (gfx1250) — matrix operand order repaired and checked by reading, not compiled, not run on hardware
//
#include <hip/hip_runtime.h>
#include <hip/hip_bf16.h>
#include <math.h>

typedef __attribute__((ext_vector_type(16))) __bf16 v16bf;
typedef __attribute__((ext_vector_type(8)))  float  v8f;

#define D 512
#define XP 520   // LDS pitch for X tile (bf16 elems), 16B-aligned, bank-conflict padded

__device__ __forceinline__ unsigned short f2bf(float v) {
    unsigned int u = __float_as_uint(v);
    u += 0x7fffu + ((u >> 16) & 1u);   // round-to-nearest-even
    return (unsigned short)(u >> 16);
}

// ---------------------------------------------------------------------------
// BatchNorm stats over batch dim (training-mode, biased variance).
// One block, 512 threads: thread d owns feature d, loops over the 512 rows.
// Produces per-feature scale/shift so bn(x) = x*scale + shift.
// idx==nullptr -> dense rows of src; else gather rows src[idx[b]].
// ---------------------------------------------------------------------------
__global__ __launch_bounds__(512)
void bn_stats_kernel(const float* __restrict__ src, const int* __restrict__ idx,
                     const float* __restrict__ gamma, const float* __restrict__ beta,
                     float* __restrict__ scale, float* __restrict__ shift) {
    int d = threadIdx.x;
    float s = 0.f, ss = 0.f;
    for (int b = 0; b < D; ++b) {
        int row = idx ? idx[b] : b;
        float v = src[(size_t)row * D + d];
        s += v; ss += v * v;
    }
    float mean = s * (1.0f / D);
    float var  = fmaxf(ss * (1.0f / D) - mean * mean, 0.f);
    float sc = gamma[d] * rsqrtf(var + 1e-5f);
    scale[d] = sc;
    shift[d] = beta[d] - mean * sc;
}

// ---------------------------------------------------------------------------
// Gather + bn0 + f32->bf16 for X; gather r rows; zero the GEMM accumulator.
// ---------------------------------------------------------------------------
__global__ __launch_bounds__(256)
void prep_kernel(const float* __restrict__ Eembed, const float* __restrict__ Rembed,
                 const int* __restrict__ e1_idx, const int* __restrict__ r_idx,
                 const float* __restrict__ scale0, const float* __restrict__ shift0,
                 unsigned short* __restrict__ Xb, float* __restrict__ Rg,
                 float* __restrict__ out1) {
    int b = blockIdx.x;
    const float* er = Eembed + (size_t)e1_idx[b] * D;
    const float* rr = Rembed + (size_t)r_idx[b] * D;
    for (int d = threadIdx.x; d < D; d += 256) {
        float x = er[d] * scale0[d] + shift0[d];
        Xb[b * D + d]  = f2bf(x);
        Rg[b * D + d]  = rr[d];
        out1[b * D + d] = 0.f;
    }
}

// ---------------------------------------------------------------------------
// Core fused contraction: out1[b,j] = sum_k r[b,k] * sum_i x[b,i] * W[k,i,j]
// GEMM view: M=512(batch) x N=512(j) x K=512*512, A[b,(k,i)] = r[b,k]*x[b,i].
// Grid: (4 N-tiles, 4 M-tiles, 32 K-chunks of 16 k-values). x-fastest dispatch
// keeps one 16-MB W k-chunk L2-resident across all 16 tiles.
// Block: 256 threads = 8 waves; wave (wrow,wcol) owns a 64x32 sub-tile
// (4x2 accumulators of v_wmma_f32_16x16x32_bf16).
// ---------------------------------------------------------------------------
__global__ __launch_bounds__(256)
void tucker_gemm_kernel(const unsigned short* __restrict__ Xb,
                        const float* __restrict__ Rg,
                        const float* __restrict__ W,
                        float* __restrict__ out1) {
    __shared__ unsigned short Xs[128 * XP];      // X tile, bf16: 130 KB
    __shared__ unsigned int   Ws[2][8 * 32 * 8]; // W stage, frag-ready, dbuf: 16 KB
    __shared__ float          Rs[128 * 16];      // r[b, kbase..kbase+15]: 8 KB

    const int tid  = threadIdx.x;
    const int lane = tid & 31;
    const int wave = tid >> 5;
    const int wrow = wave >> 2;        // 0..1 -> 64-row strip
    const int wcol = wave & 3;         // 0..3 -> 32-col strip
    const int Nbase = blockIdx.x * 128;
    const int Mbase = blockIdx.y * 128;
    const int kbase = blockIdx.z * 16;

    // ---- stage X tile (128 x 512 bf16) via async global->LDS DMA ----
    {
        unsigned xs_base = (unsigned)(size_t)Xs;           // wave-relative LDS offset
        for (int idx = tid; idx < 128 * (D / 8); idx += 256) {
            int row = idx >> 6, c = idx & 63;
            unsigned lds_off = xs_base + row * (XP * 2) + c * 16;
            const uint4* src = (const uint4*)(Xb + (size_t)(Mbase + row) * D) + c;
            asm volatile("global_load_async_to_lds_b128 %0, %1, off"
                         :: "v"(lds_off), "v"(src) : "memory");
        }
    }
    // ---- stage r tile (128 x 16 f32) ----
    for (int idx = tid; idx < 128 * 16; idx += 256) {
        int row = idx >> 4, kk = idx & 15;
        Rs[idx] = Rg[(size_t)(Mbase + row) * D + kbase + kk];
    }

    // lane's fixed column for W staging (fragment-ready layout)
    const int jgrp = wave;                                 // 0..7 (16 cols each)
    const int jloc = Nbase + jgrp * 16 + (lane & 15);

    v8f acc[4][2] = {};
    unsigned int rsc[4];

    // Load one 32(K) x 128(N) chunk of W[k] straight into per-lane frag regs.
    auto LOADW = [&](int s, float* f) {
        int k  = kbase + (s >> 4);
        int i0 = ((s & 15) << 5) + ((lane >> 4) << 4);
        const float* wp = W + (((size_t)k << 18) + ((size_t)i0 << 9)) + jloc;
        __builtin_prefetch(wp + (32 << 9), 0, 0);          // next i-chunk -> global_prefetch_b8
#pragma unroll
        for (int g = 0; g < 16; ++g) f[g] = wp[(size_t)g << 9];
    };
    // Pack to bf16 pairs along K (v_cvt_pk_bf16_f32) and store lane-major
    // (dense B 32x16 bf16 fragment layout).
    auto STOREW = [&](int s, const float* f) {
        unsigned int* dst = &Ws[s & 1][(jgrp * 32 + lane) * 8];
        uint4 q0, q1;
        unsigned int t[8];
#pragma unroll
        for (int g = 0; g < 8; ++g) {
            unsigned int pk;
            asm("v_cvt_pk_bf16_f32 %0, %1, %2"
                : "=v"(pk) : "v"(f[2 * g]), "v"(f[2 * g + 1]));
            t[g] = pk;
        }
        q0.x = t[0]; q0.y = t[1]; q0.z = t[2]; q0.w = t[3];
        q1.x = t[4]; q1.y = t[5]; q1.z = t[6]; q1.w = t[7];
        ((uint4*)dst)[0] = q0;
        ((uint4*)dst)[1] = q1;
    };
    // Per-lane r[b,k] scale, replicated into a packed-bf16 dword.
    auto RSCALE = [&](int kk) {
#pragma unroll
        for (int m = 0; m < 4; ++m) {
            float rv = Rs[(wrow * 64 + m * 16 + (lane & 15)) * 16 + kk];
            unsigned int hb = (unsigned int)f2bf(rv);
            rsc[m] = hb | (hb << 16);
        }
    };
    auto COMPUTE = [&](int s) {
        int ib = (s & 15) << 5;                            // K offset within X row
        v16bf bf[2];
#pragma unroll
        for (int n = 0; n < 2; ++n) {
            const uint4* bp = (const uint4*)&Ws[s & 1][((wcol * 2 + n) * 32 + lane) * 8];
            ((uint4*)&bf[n])[0] = bp[0];
            ((uint4*)&bf[n])[1] = bp[1];
        }
#pragma unroll
        for (int m = 0; m < 4; ++m) {
            int row = wrow * 64 + m * 16 + (lane & 15);
            const unsigned short* xr = Xs + (size_t)row * XP + ib + ((lane >> 4) << 4);
            v16bf af;
            ((uint4*)&af)[0] = *(const uint4*)(xr);
            ((uint4*)&af)[1] = *(const uint4*)(xr + 8);
            // fold r[b,k] into the A fragment: packed bf16 multiply (VOP3P)
#pragma unroll
            for (int q = 0; q < 8; ++q) {
                unsigned int dv;
                asm("v_pk_mul_bf16 %0, %1, %2"
                    : "=v"(dv)
                    : "v"(((unsigned int*)&af)[q]), "v"(rsc[m]));
                ((unsigned int*)&af)[q] = dv;
            }
#pragma unroll
            for (int n = 0; n < 2; ++n)
                acc[m][n] = __builtin_amdgcn_wmma_f32_16x16x32_bf16(
                    false, af, false, bf[n], (short)0, acc[m][n], false, false);
        }
    };

    // ---- software-pipelined main loop: 16 k-values x 16 K-steps = 256 stages
    float cur[16];
    LOADW(0, cur);
    STOREW(0, cur);
    // X-tile DMA must have landed before the first barrier releases compute.
    asm volatile("s_wait_asynccnt 0x0" ::: "memory");
    for (int s = 0; s < 256; ++s) {
        __syncthreads();
        float nxt[16];
        if (s + 1 < 256) LOADW(s + 1, nxt);                // overlap with compute below
        if ((s & 15) == 0) RSCALE(s >> 4);
        COMPUTE(s);
        if (s + 1 < 256) STOREW(s + 1, nxt);
    }

    // ---- split-K accumulate into f32 intermediate ----
#pragma unroll
    for (int m = 0; m < 4; ++m) {
#pragma unroll
        for (int n = 0; n < 2; ++n) {
            int col   = Nbase + wcol * 32 + n * 16 + (lane & 15);
            int rbase = Mbase + wrow * 64 + m * 16 + ((lane >> 4) << 3);
#pragma unroll
            for (int g = 0; g < 8; ++g)
                atomicAdd(&out1[(size_t)(rbase + g) * D + col], acc[m][n][g]);
        }
    }
}

// ---------------------------------------------------------------------------
// bn1 + gathers of e2p/e2n + row dots + sigmoid. One block per batch row.
// ---------------------------------------------------------------------------
__global__ __launch_bounds__(256)
void final_kernel(const float* __restrict__ out1,
                  const float* __restrict__ scale1, const float* __restrict__ shift1,
                  const float* __restrict__ Eembed,
                  const int* __restrict__ e2p_idx, const int* __restrict__ e2n_idx,
                  float* __restrict__ outp) {
    __shared__ float sp[256], sn[256];
    int b = blockIdx.x;
    const float* xr = out1 + (size_t)b * D;
    const float* pr = Eembed + (size_t)e2p_idx[b] * D;
    const float* nr = Eembed + (size_t)e2n_idx[b] * D;
    float ap = 0.f, an = 0.f;
    for (int d = threadIdx.x; d < D; d += 256) {
        float xn = xr[d] * scale1[d] + shift1[d];
        ap += xn * pr[d];
        an += xn * nr[d];
    }
    sp[threadIdx.x] = ap; sn[threadIdx.x] = an;
    __syncthreads();
    for (int off = 128; off > 0; off >>= 1) {
        if (threadIdx.x < off) {
            sp[threadIdx.x] += sp[threadIdx.x + off];
            sn[threadIdx.x] += sn[threadIdx.x + off];
        }
        __syncthreads();
    }
    if (threadIdx.x == 0) {
        outp[b]     = 1.f / (1.f + expf(-sp[0]));
        outp[D + b] = 1.f / (1.f + expf(-sn[0]));
    }
}

// ---------------------------------------------------------------------------
extern "C" void kernel_launch(void* const* d_in, const int* in_sizes, int n_in,
                              void* d_out, int out_size, void* d_ws, size_t ws_size,
                              hipStream_t stream) {
    const int*   e1_idx  = (const int*)d_in[0];
    const int*   r_idx   = (const int*)d_in[1];
    const int*   e2p_idx = (const int*)d_in[2];
    const int*   e2n_idx = (const int*)d_in[3];
    const float* Eembed  = (const float*)d_in[4];
    const float* Rembed  = (const float*)d_in[5];
    const float* W       = (const float*)d_in[6];
    const float* bn0_g   = (const float*)d_in[7];
    const float* bn0_b   = (const float*)d_in[8];
    const float* bn1_g   = (const float*)d_in[9];
    const float* bn1_b   = (const float*)d_in[10];

    // workspace carve-up
    float*          scale0 = (float*)d_ws;               // 512
    float*          shift0 = scale0 + 512;               // 512
    float*          scale1 = scale0 + 1024;              // 512
    float*          shift1 = scale0 + 1536;              // 512
    unsigned short* Xb     = (unsigned short*)((char*)d_ws + 8192);          // 512 KB
    float*          Rg     = (float*)((char*)d_ws + 8192 + D * D * 2);       // 1 MB
    float*          out1   = Rg + (size_t)D * D;                             // 1 MB

    bn_stats_kernel<<<1, 512, 0, stream>>>(Eembed, e1_idx, bn0_g, bn0_b, scale0, shift0);
    prep_kernel<<<D, 256, 0, stream>>>(Eembed, Rembed, e1_idx, r_idx,
                                       scale0, shift0, Xb, Rg, out1);
    tucker_gemm_kernel<<<dim3(4, 4, 32), 256, 0, stream>>>(Xb, Rg, W, out1);
    bn_stats_kernel<<<1, 512, 0, stream>>>(out1, nullptr, bn1_g, bn1_b, scale1, shift1);
    final_kernel<<<D, 256, 0, stream>>>(out1, scale1, shift1, Eembed,
                                        e2p_idx, e2n_idx, (float*)d_out);
}
